// RelAttention_48619029791172
// MI455X (gfx1250) — matrix operand order repaired and checked by reading, not compiled, not run on hardware
//
#include <hip/hip_runtime.h>
#include <hip/hip_bf16.h>
#include <math.h>

typedef __bf16 bf16;
typedef __attribute__((ext_vector_type(16))) __bf16 v16bf;
typedef __attribute__((ext_vector_type(8)))  __bf16 v8bf;
typedef __attribute__((ext_vector_type(8)))  float  v8f;
typedef __attribute__((ext_vector_type(4)))  int    v4i;

#define BM 256
#define BN 128
#define BK 32
#define LDSS 40   // padded LDS row stride in bf16 elems (80B: conflict-free b128 reads)

enum { CM_F32 = 0, CM_BF16_BIAS = 1, CM_BF16T_BIAS = 2 };

#if __has_builtin(__builtin_amdgcn_global_load_async_to_lds_b128) && \
    __has_builtin(__builtin_amdgcn_s_wait_asynccnt)
#define HAVE_ASYNC_LDS 1
#else
#define HAVE_ASYNC_LDS 0
#endif

#if HAVE_ASYNC_LDS
typedef __attribute__((address_space(1))) v4i as1_v4i;
typedef __attribute__((address_space(3))) v4i as3_v4i;
__device__ __forceinline__ void async_b128(const char* g, bf16* l) {
  // LDS generic pointer keeps the byte offset in the low 32 bits (aperture rules).
  __builtin_amdgcn_global_load_async_to_lds_b128(
      (as1_v4i*)(unsigned long long)g,
      (as3_v4i*)(unsigned int)(unsigned long long)l, 0, 0);
}
__device__ __forceinline__ void async_wait0() {
  __builtin_amdgcn_s_wait_asynccnt(0);
}
#else
__device__ __forceinline__ void async_b128(const char*, bf16*) {}
__device__ __forceinline__ void async_wait0() {}
#endif

// C = A * B^T.  A: [M,K] row-major (f32 or bf16).  Bsrc: [N,K] row-major (same elem type).
// Block 256x128, 8 waves in 4(M) x 2(N) grid, each wave owns 64x64 (4x4 16x16 tiles).
template<bool SRCF32, int CMODE>
__global__ __launch_bounds__(256) void gemm_nt(
    const void* __restrict__ Av, const void* __restrict__ Bv,
    const float* __restrict__ bias, void* __restrict__ Cv,
    int M, int N, int K,
    long long strideA, long long strideB, long long strideC, int S)
{
  __shared__ __align__(16) bf16 sA[2][BM * LDSS];
  __shared__ __align__(16) bf16 sB[2][BN * LDSS];

  const int tid  = threadIdx.x;
  const int brow = tid >> 1;          // 0..127: B-tile row for this thread
  const int bcol = (tid & 1) * 16;    // 0 or 16

  const int esz = SRCF32 ? 4 : 2;
  const char* Abase = (const char*)Av + (size_t)blockIdx.z * (size_t)strideA * esz;
  const char* Bbase = (const char*)Bv + (size_t)blockIdx.z * (size_t)strideB * esz;

  const int blockM = blockIdx.y * BM;
  const int blockN = blockIdx.x * BN;

  const int lane  = tid & 31;
  const int wave  = tid >> 5;
  const int waveM = (wave >> 1) * 64;   // 4 waves along M
  const int waveN = (wave & 1) * 64;    // 2 waves along N
  const int half  = lane >> 4;
  const int l16   = lane & 15;

  constexpr bool USE_ASYNC = (!SRCF32) && (HAVE_ASYNC_LDS != 0);

  bf16 rA[32], rB[16];

  // ---- loaders: A tile is 256 rows x 32 elems (1 row/thread); B is 128 x 32 (half row/thread)
  auto fetchA = [&](int k0) {
    const size_t goff = (size_t)(blockM + tid) * K + k0;
    if (SRCF32) {
      const float4* p = (const float4*)(Abase + goff * 4);
      float t[32] __attribute__((aligned(16)));
#pragma unroll
      for (int c = 0; c < 8; ++c) *(float4*)(t + 4 * c) = p[c];
#pragma unroll
      for (int j = 0; j < 32; ++j) rA[j] = (bf16)t[j];
    } else {
      const v8bf* p = (const v8bf*)(Abase + goff * 2);
#pragma unroll
      for (int c = 0; c < 4; ++c) {
        v8bf v = p[c];
#pragma unroll
        for (int j = 0; j < 8; ++j) rA[8 * c + j] = v[j];
      }
    }
  };
  auto stashA = [&](bf16* s) {
    bf16* dst = s + tid * LDSS;
#pragma unroll
    for (int c = 0; c < 4; ++c) {
      v8bf v;
#pragma unroll
      for (int j = 0; j < 8; ++j) v[j] = rA[8 * c + j];
      *(v8bf*)(dst + 8 * c) = v;
    }
  };
  auto asyncA = [&](int k0, bf16* s) {
    const char* g = Abase + ((size_t)(blockM + tid) * K + k0) * 2;
    bf16* l = s + tid * LDSS;
#pragma unroll
    for (int c = 0; c < 4; ++c) async_b128(g + 16 * c, l + 8 * c);
  };

  auto fetchB = [&](int k0) {
    const size_t goff = (size_t)(blockN + brow) * K + k0 + bcol;
    if (SRCF32) {
      const float4* p = (const float4*)(Bbase + goff * 4);
      float t[16] __attribute__((aligned(16)));
#pragma unroll
      for (int c = 0; c < 4; ++c) *(float4*)(t + 4 * c) = p[c];
#pragma unroll
      for (int j = 0; j < 16; ++j) rB[j] = (bf16)t[j];
    } else {
      const v8bf* p = (const v8bf*)(Bbase + goff * 2);
#pragma unroll
      for (int c = 0; c < 2; ++c) {
        v8bf v = p[c];
#pragma unroll
        for (int j = 0; j < 8; ++j) rB[8 * c + j] = v[j];
      }
    }
  };
  auto stashB = [&](bf16* s) {
    bf16* dst = s + brow * LDSS + bcol;
#pragma unroll
    for (int c = 0; c < 2; ++c) {
      v8bf v;
#pragma unroll
      for (int j = 0; j < 8; ++j) v[j] = rB[8 * c + j];
      *(v8bf*)(dst + 8 * c) = v;
    }
  };
  auto asyncB = [&](int k0, bf16* s) {
    const char* g = Bbase + ((size_t)(blockN + brow) * K + k0 + bcol) * 2;
    bf16* l = s + brow * LDSS + bcol;
#pragma unroll
    for (int c = 0; c < 2; ++c) async_b128(g + 16 * c, l + 8 * c);
  };

  v8f acc[4][4];
  const v8f vzero = {0.f, 0.f, 0.f, 0.f, 0.f, 0.f, 0.f, 0.f};
#pragma unroll
  for (int i = 0; i < 4; ++i)
#pragma unroll
    for (int j = 0; j < 4; ++j) acc[i][j] = vzero;

  const int KT = K / BK;

  if (USE_ASYNC) {
    asyncA(0, sA[0]);
    asyncB(0, sB[0]);
    async_wait0();
  } else {
    fetchA(0); fetchB(0);
    stashA(sA[0]); stashB(sB[0]);
  }
  __syncthreads();

  for (int kt = 0; kt < KT; ++kt) {
    if (kt + 1 < KT) {
      if (USE_ASYNC) {
        asyncA((kt + 1) * BK, sA[(kt + 1) & 1]);
        asyncB((kt + 1) * BK, sB[(kt + 1) & 1]);
      } else {
        fetchA((kt + 1) * BK);
        fetchB((kt + 1) * BK);
      }
    }
    if (kt + 2 < KT) {
      __builtin_prefetch(Abase + ((size_t)(blockM + tid) * K + (kt + 2) * BK) * esz, 0, 1);
      __builtin_prefetch(Bbase + ((size_t)(blockN + brow) * K + (kt + 2) * BK + bcol) * esz, 0, 1);
    }

    const bf16* cA = sA[kt & 1];
    const bf16* cB = sB[kt & 1];
    // B frag: lanes 0-15 -> K 0..15, lanes 16-31 -> K 16..31; N = lane&15 (Bsrc stored [N,K])
    v16bf b[4];
#pragma unroll
    for (int j = 0; j < 4; ++j) {
      const bf16* p = cB + (waveN + j * 16 + l16) * LDSS + half * 16;
      v8bf x0 = *(const v8bf*)p;
      v8bf x1 = *(const v8bf*)(p + 8);
#pragma unroll
      for (int e = 0; e < 8; ++e) { b[j][e] = x0[e]; b[j][8 + e] = x1[e]; }
    }
    // A frag: lanes 0-15 -> K 0..15, lanes 16-31 -> K 16..31; M = lane&15
#pragma unroll
    for (int i = 0; i < 4; ++i) {
      const bf16* p = cA + (waveM + i * 16 + l16) * LDSS + half * 16;
      v8bf x0 = *(const v8bf*)p;
      v8bf x1 = *(const v8bf*)(p + 8);
      v16bf a;
#pragma unroll
      for (int e = 0; e < 8; ++e) { a[e] = x0[e]; a[8 + e] = x1[e]; }
#pragma unroll
      for (int j = 0; j < 4; ++j)
        acc[i][j] = __builtin_amdgcn_wmma_f32_16x16x32_bf16(
            false, a, false, b[j], (short)0, acc[i][j], false, false);
    }

    if (kt + 1 < KT) {
      if (USE_ASYNC) async_wait0();
      else { stashA(sA[(kt + 1) & 1]); stashB(sB[(kt + 1) & 1]); }
    }
    __syncthreads();
  }

  // Epilogue.  Acc VGPR r: lanes 0-15 -> M=r, lanes 16-31 -> M=r+8; N = lane&15.
#pragma unroll
  for (int i = 0; i < 4; ++i) {
#pragma unroll
    for (int j = 0; j < 4; ++j) {
      const int mbase = blockM + waveM + i * 16 + half * 8;
      const int ncol  = blockN + waveN + j * 16 + l16;
      float badd = 0.f;
      if (CMODE != CM_F32) badd = bias[ncol];
      if (CMODE == CM_BF16T_BIAS) {
        // rows mbase..mbase+7 are consecutive t in the same batch -> one packed 16B store
        const int bidx = mbase / S;
        const int t0   = mbase - bidx * S;
        v8bf pk;
#pragma unroll
        for (int r = 0; r < 8; ++r) pk[r] = (bf16)(acc[i][j][r] + badd);
        *(v8bf*)&((bf16*)Cv)[((size_t)bidx * N + ncol) * S + t0] = pk;
      } else {
#pragma unroll
        for (int r = 0; r < 8; ++r) {
          const int mrow = mbase + r;
          const float v  = acc[i][j][r];
          if (CMODE == CM_F32) {
            ((float*)Cv)[(size_t)blockIdx.z * (size_t)strideC + (size_t)mrow * N + ncol] = v;
          } else {
            ((bf16*)Cv)[(size_t)mrow * N + ncol] = (bf16)(v + badd);
          }
        }
      }
    }
  }
}

// logits = att*scale + br*pr + bp*pp - bi*pi; softmax over the row; write bf16 weights.
template<int NITER>
__global__ __launch_bounds__(256) void softmax_phi_fast(
    const float* __restrict__ att,
    const float* __restrict__ pr, const float* __restrict__ pp, const float* __restrict__ pi,
    const float* __restrict__ brp, const float* __restrict__ bpp, const float* __restrict__ bip,
    bf16* __restrict__ w, int S, float scale)
{
  __shared__ float red[256];
  const size_t off = (size_t)blockIdx.x * S;
  const int tid = threadIdx.x;
  const float br = *brp, bp = *bpp, bi = *bip;

  float lv[NITER];
  float mx = -3.402823466e38f;
#pragma unroll
  for (int it = 0; it < NITER; ++it) {
    const size_t c = off + tid + it * 256;
    const float v = att[c] * scale + br * pr[c] + bp * pp[c] - bi * pi[c];
    lv[it] = v;
    mx = fmaxf(mx, v);
  }
  red[tid] = mx;
  __syncthreads();
  for (int s2 = 128; s2 > 0; s2 >>= 1) {
    if (tid < s2) red[tid] = fmaxf(red[tid], red[tid + s2]);
    __syncthreads();
  }
  mx = red[0];
  __syncthreads();
  float sum = 0.f;
#pragma unroll
  for (int it = 0; it < NITER; ++it) { lv[it] = __expf(lv[it] - mx); sum += lv[it]; }
  red[tid] = sum;
  __syncthreads();
  for (int s2 = 128; s2 > 0; s2 >>= 1) {
    if (tid < s2) red[tid] += red[tid + s2];
    __syncthreads();
  }
  const float inv = 1.f / red[0];
#pragma unroll
  for (int it = 0; it < NITER; ++it)
    w[off + tid + it * 256] = (bf16)(lv[it] * inv);
}

__global__ __launch_bounds__(256) void softmax_phi_gen(
    const float* __restrict__ att,
    const float* __restrict__ pr, const float* __restrict__ pp, const float* __restrict__ pi,
    const float* __restrict__ brp, const float* __restrict__ bpp, const float* __restrict__ bip,
    bf16* __restrict__ w, int S, float scale)
{
  __shared__ float red[256];
  const size_t off = (size_t)blockIdx.x * S;
  const int tid = threadIdx.x;
  const float br = *brp, bp = *bpp, bi = *bip;
  float mx = -3.402823466e38f;
  for (int c = tid; c < S; c += 256) {
    float v = att[off + c] * scale + br * pr[off + c] + bp * pp[off + c] - bi * pi[off + c];
    mx = fmaxf(mx, v);
  }
  red[tid] = mx; __syncthreads();
  for (int s2 = 128; s2 > 0; s2 >>= 1) {
    if (tid < s2) red[tid] = fmaxf(red[tid], red[tid + s2]);
    __syncthreads();
  }
  mx = red[0]; __syncthreads();
  float sum = 0.f;
  for (int c = tid; c < S; c += 256) {
    float v = att[off + c] * scale + br * pr[off + c] + bp * pp[off + c] - bi * pi[off + c];
    sum += __expf(v - mx);
  }
  red[tid] = sum; __syncthreads();
  for (int s2 = 128; s2 > 0; s2 >>= 1) {
    if (tid < s2) red[tid] += red[tid + s2];
    __syncthreads();
  }
  const float inv = 1.f / red[0];
  for (int c = tid; c < S; c += 256) {
    float v = att[off + c] * scale + br * pr[off + c] + bp * pp[off + c] - bi * pi[off + c];
    w[off + c] = (bf16)(__expf(v - mx) * inv);
  }
}

extern "C" void kernel_launch(void* const* d_in, const int* in_sizes, int n_in,
                              void* d_out, int out_size, void* d_ws, size_t ws_size,
                              hipStream_t stream) {
  const float* q_in  = (const float*)d_in[0];
  const float* k_in  = (const float*)d_in[1];
  const float* v_in  = (const float*)d_in[2];
  const float* phi_r = (const float*)d_in[3];
  const float* phi_p = (const float*)d_in[4];
  const float* phi_i = (const float*)d_in[5];
  const float* Wq    = (const float*)d_in[6];
  const float* bq    = (const float*)d_in[7];
  const float* Wk    = (const float*)d_in[8];
  const float* bk    = (const float*)d_in[9];
  const float* Wv    = (const float*)d_in[10];
  const float* bv    = (const float*)d_in[11];
  const float* be_r  = (const float*)d_in[12];
  const float* be_p  = (const float*)d_in[13];
  const float* be_i  = (const float*)d_in[14];

  const int d      = in_sizes[7];                            // len(bq) == d
  const int BSrows = in_sizes[0] / d;                        // B*S
  const int S      = (int)((long long)in_sizes[3] / BSrows); // B*S*S / (B*S)
  const int B      = BSrows / S;

  // Workspace layout
  bf16*  Qbf = (bf16*)d_ws;                             // [B*S, d]
  bf16*  Kbf = Qbf + (size_t)BSrows * d;                // [B*S, d]  ([N,K] form for scores)
  bf16*  Vt  = Kbf + (size_t)BSrows * d;                // [B, d, S] ([N,K] form for output)
  float* att = (float*)(Vt + (size_t)BSrows * d);       // [B, S, S]
  bf16*  wbf = (bf16*)(att + (size_t)BSrows * S);       // [B, S, S]

  dim3 blk(256);

  // Projections: C = X * W^T + b  (W row-major [n,k] is already the [N,K] operand)
  gemm_nt<true, CM_BF16_BIAS><<<dim3(d / BN, BSrows / BM, 1), blk, 0, stream>>>(
      q_in, Wq, bq, Qbf, BSrows, d, d, 0, 0, 0, S);
  gemm_nt<true, CM_BF16_BIAS><<<dim3(d / BN, BSrows / BM, 1), blk, 0, stream>>>(
      k_in, Wk, bk, Kbf, BSrows, d, d, 0, 0, 0, S);
  gemm_nt<true, CM_BF16T_BIAS><<<dim3(d / BN, BSrows / BM, 1), blk, 0, stream>>>(
      v_in, Wv, bv, Vt, BSrows, d, d, 0, 0, 0, S);

  // Scores: att[b] = Q[b] * K[b]^T  (scale folded into softmax)
  gemm_nt<false, CM_F32><<<dim3(S / BN, S / BM, B), blk, 0, stream>>>(
      Qbf, Kbf, nullptr, att, S, S, d,
      (long long)S * d, (long long)S * d, (long long)S * S, S);

  // Softmax with phi terms
  const float scale = 1.0f / sqrtf((float)d);
  if (S % 256 == 0 && S / 256 == 8)
    softmax_phi_fast<8><<<BSrows, blk, 0, stream>>>(att, phi_r, phi_p, phi_i,
                                                    be_r, be_p, be_i, wbf, S, scale);
  else if (S % 256 == 0 && S / 256 == 4)
    softmax_phi_fast<4><<<BSrows, blk, 0, stream>>>(att, phi_r, phi_p, phi_i,
                                                    be_r, be_p, be_i, wbf, S, scale);
  else if (S % 256 == 0 && S / 256 == 16)
    softmax_phi_fast<16><<<BSrows, blk, 0, stream>>>(att, phi_r, phi_p, phi_i,
                                                     be_r, be_p, be_i, wbf, S, scale);
  else
    softmax_phi_gen<<<BSrows, blk, 0, stream>>>(att, phi_r, phi_p, phi_i,
                                                be_r, be_p, be_i, wbf, S, scale);

  // Output: out[b] = w[b] * V[b]  (Vt is the [N,K] operand)
  gemm_nt<false, CM_F32><<<dim3(d / BN, S / BM, B), blk, 0, stream>>>(
      wbf, Vt, nullptr, (float*)d_out, S, d, S,
      (long long)S * S, (long long)d * S, (long long)S * d, S);
}
